// SpatialInteraction_8126078124707
// MI455X (gfx1250) — matrix operand order repaired and checked by reading, not compiled, not run on hardware
//
#include <hip/hip_runtime.h>

// ---------------------------------------------------------------------------
// SpatialInteraction: out[b] = softmax(X[b] X[b]^T) X[b]
//   X: [16, 1024, 64*64] f32 ; GEMMs run as bf16 WMMA (v_wmma_f32_16x16x32_bf16)
//   with f32 accumulation; softmax in f32.
// Pipeline: double-buffered LDS tiles, register-staged global prefetch so the
// f32 global loads overlap the WMMA block; one barrier per K-step.
// ---------------------------------------------------------------------------

typedef __attribute__((ext_vector_type(8)))  __bf16 bf16x8;
typedef __attribute__((ext_vector_type(16))) __bf16 bf16x16;
typedef __attribute__((ext_vector_type(8)))  float  f32x8;

#define BATCH 16
#define CDIM  1024
#define HW    4096
#define LDP   40        // padded LDS pitch in ushorts (80B = 20 banks)

__device__ __forceinline__ unsigned short f32_to_bf16_rne(float f) {
  unsigned int u = __float_as_uint(f);
  u += 0x7FFFu + ((u >> 16) & 1u);          // round-to-nearest-even
  return (unsigned short)(u >> 16);
}

__device__ __forceinline__ bf16x16 make_frag(const unsigned short* p0,
                                             const unsigned short* p1) {
  bf16x8 lo = *reinterpret_cast<const bf16x8*>(p0);
  bf16x8 hi = *reinterpret_cast<const bf16x8*>(p1);
  return __builtin_shufflevector(lo, hi, 0,1,2,3,4,5,6,7,8,9,10,11,12,13,14,15);
}

// ---------------------------------------------------------------------------
// Kernel 1: S[b] = X[b] (1024x4096) * X[b]^T   (f32 accum via bf16 WMMA)
// Block tile 128x128, K in steps of 32; 8 waves = 2(M) x 4(N), each wave owns
// 4x2 16x16 accumulator tiles. B = X^T, so both operands stage from row-major
// X with K contiguous.
// ---------------------------------------------------------------------------
__global__ __launch_bounds__(256)
void gemm1_scores(const float* __restrict__ x, float* __restrict__ S) {
  __shared__ unsigned short sA[2][128 * LDP];
  __shared__ unsigned short sB[2][128 * LDP];

  const int b  = blockIdx.z;
  const int m0 = blockIdx.y * 128;
  const int n0 = blockIdx.x * 128;
  const float* xb = x + (size_t)b * CDIM * HW;

  const int t      = threadIdx.x;
  const int lane   = t & 31;
  const int lane16 = lane & 15;
  const int hp     = lane >> 4;       // half-wave select
  const int wave   = t >> 5;
  const int wM     = wave & 1;        // 0..1 -> 64 rows each
  const int wN     = wave >> 1;       // 0..3 -> 32 cols each

  const int sr = t >> 3;              // staging row within 32-row pass
  const int sc = (t & 7) * 4;         // staging col (4 floats)

  f32x8 acc[4][2] = {};
  float4 ra[4], rb[4];

  auto load_g = [&](int k0) {
    __builtin_prefetch(xb + (size_t)(m0 + sr) * HW + k0 + 32 + sc, 0, 1);
    __builtin_prefetch(xb + (size_t)(n0 + sr) * HW + k0 + 32 + sc, 0, 1);
#pragma unroll
    for (int p = 0; p < 4; ++p) {
      const int r = p * 32 + sr;
      ra[p] = *reinterpret_cast<const float4*>(xb + (size_t)(m0 + r) * HW + k0 + sc);
      rb[p] = *reinterpret_cast<const float4*>(xb + (size_t)(n0 + r) * HW + k0 + sc);
    }
  };
  auto store_l = [&](int buf) {
#pragma unroll
    for (int p = 0; p < 4; ++p) {
      const int r = p * 32 + sr;
      unsigned short* a = &sA[buf][r * LDP + sc];
      unsigned short* bq = &sB[buf][r * LDP + sc];
      a[0] = f32_to_bf16_rne(ra[p].x); a[1] = f32_to_bf16_rne(ra[p].y);
      a[2] = f32_to_bf16_rne(ra[p].z); a[3] = f32_to_bf16_rne(ra[p].w);
      bq[0] = f32_to_bf16_rne(rb[p].x); bq[1] = f32_to_bf16_rne(rb[p].y);
      bq[2] = f32_to_bf16_rne(rb[p].z); bq[3] = f32_to_bf16_rne(rb[p].w);
    }
  };

  load_g(0);
  store_l(0);
  __syncthreads();

  int cur = 0;
  for (int k0 = 0; k0 < HW; k0 += 32) {
    const int nxt = k0 + 32;
    if (nxt < HW) load_g(nxt);        // in flight across the WMMA block

    bf16x16 af[4], bf[2];
#pragma unroll
    for (int mt = 0; mt < 4; ++mt) {  // A frag: 16B chunks at bytes 16*hp, 32+16*hp
      const unsigned short* pr = &sA[cur][(wM * 64 + mt * 16 + lane16) * LDP + hp * 8];
      af[mt] = make_frag(pr, pr + 16);
    }
#pragma unroll
    for (int nt = 0; nt < 2; ++nt) {  // B frag: 16B chunks at bytes 16*hp, 32+16*hp
      const unsigned short* pr = &sB[cur][(wN * 32 + nt * 16 + lane16) * LDP + hp * 8];
      bf[nt] = make_frag(pr, pr + 16);
    }
#pragma unroll
    for (int mt = 0; mt < 4; ++mt)
#pragma unroll
      for (int nt = 0; nt < 2; ++nt)
        acc[mt][nt] = __builtin_amdgcn_wmma_f32_16x16x32_bf16(
            false, af[mt], false, bf[nt], (short)0, acc[mt][nt], false, false);

    if (nxt < HW) {
      store_l(cur ^ 1);               // other buffer: safe vs. concurrent readers
      __syncthreads();                // single barrier per K-step
      cur ^= 1;
    }
  }

  float* Sb = S + (size_t)b * CDIM * CDIM;
#pragma unroll
  for (int mt = 0; mt < 4; ++mt)
#pragma unroll
    for (int nt = 0; nt < 2; ++nt)
#pragma unroll
      for (int v = 0; v < 8; ++v) {
        const int m = m0 + wM * 64 + mt * 16 + v + 8 * hp;
        const int n = n0 + wN * 32 + nt * 16 + lane16;
        Sb[(size_t)m * CDIM + n] = acc[mt][nt][v];
      }
}

// ---------------------------------------------------------------------------
// Kernel 2: row softmax (f32) -> bf16 attention. One 256-thread WG per row.
// ---------------------------------------------------------------------------
__global__ __launch_bounds__(256)
void softmax_rows(const float* __restrict__ S, unsigned short* __restrict__ P) {
  __shared__ float red[256];
  const int row = blockIdx.x;
  const int t   = threadIdx.x;
  const float* s = S + (size_t)row * CDIM;

  float v[4];
  float m = -__builtin_inff();
#pragma unroll
  for (int i = 0; i < 4; ++i) { v[i] = s[t + 256 * i]; m = fmaxf(m, v[i]); }
  red[t] = m; __syncthreads();
  for (int off = 128; off > 0; off >>= 1) {
    if (t < off) red[t] = fmaxf(red[t], red[t + off]);
    __syncthreads();
  }
  m = red[0]; __syncthreads();

  float sum = 0.f;
#pragma unroll
  for (int i = 0; i < 4; ++i) { v[i] = __expf(v[i] - m); sum += v[i]; }
  red[t] = sum; __syncthreads();
  for (int off = 128; off > 0; off >>= 1) {
    if (t < off) red[t] += red[t + off];
    __syncthreads();
  }
  const float inv = 1.0f / red[0];

  unsigned short* p = P + (size_t)row * CDIM;
#pragma unroll
  for (int i = 0; i < 4; ++i) p[t + 256 * i] = f32_to_bf16_rne(v[i] * inv);
}

// ---------------------------------------------------------------------------
// Kernel 3: out[b] = P[b] (1024x1024 bf16) * X[b] (1024x4096), K = 1024.
// B = X has K along c (row stride 4096): stage the 32(k) x 128(n) f32 slice
// into LDS transposed (sB[n_local][k_local]) so B-frags are contiguous-K.
// Same double-buffered pipeline as kernel 1.
// ---------------------------------------------------------------------------
__global__ __launch_bounds__(256)
void gemm2_out(const unsigned short* __restrict__ P, const float* __restrict__ x,
               float* __restrict__ out) {
  __shared__ unsigned short sA[2][128 * LDP];
  __shared__ unsigned short sB[2][128 * LDP];

  const int b  = blockIdx.z;
  const int m0 = blockIdx.y * 128;   // c rows
  const int n0 = blockIdx.x * 128;   // hw cols
  const unsigned short* Pb = P + (size_t)b * CDIM * CDIM;
  const float* xb = x + (size_t)b * CDIM * HW;

  const int t      = threadIdx.x;
  const int lane   = t & 31;
  const int lane16 = lane & 15;
  const int hp     = lane >> 4;
  const int wave   = t >> 5;
  const int wM     = wave & 1;
  const int wN     = wave >> 1;

  const int ar  = t >> 2;            // A staging row within 64-row pass
  const int ac  = (t & 3) * 8;       // A staging col (8 bf16 = 16B)
  const int bk  = t >> 5;            // B staging k within 8-row pass
  const int bn  = (t & 31) * 4;      // B staging n (4 floats)

  f32x8 acc[4][2] = {};
  uint4  rap[2];
  float4 rbp[4];

  auto load_g = [&](int k0) {
    __builtin_prefetch(xb + (size_t)(k0 + 32 + bk) * HW + n0 + bn, 0, 1);
#pragma unroll
    for (int p = 0; p < 2; ++p) {    // A: bf16 rows of attn (direct 16B copies)
      const int r = p * 64 + ar;
      rap[p] = *reinterpret_cast<const uint4*>(Pb + (size_t)(m0 + r) * CDIM + k0 + ac);
    }
#pragma unroll
    for (int p = 0; p < 4; ++p) {    // B: f32 32(k) x 128(n) slice
      const int kl = p * 8 + bk;
      rbp[p] = *reinterpret_cast<const float4*>(xb + (size_t)(k0 + kl) * HW + n0 + bn);
    }
  };
  auto store_l = [&](int buf) {
#pragma unroll
    for (int p = 0; p < 2; ++p) {
      const int r = p * 64 + ar;
      *reinterpret_cast<uint4*>(&sA[buf][r * LDP + ac]) = rap[p];
    }
#pragma unroll
    for (int p = 0; p < 4; ++p) {    // transpose into sB[n][k]
      const int kl = p * 8 + bk;
      sB[buf][(bn + 0) * LDP + kl] = f32_to_bf16_rne(rbp[p].x);
      sB[buf][(bn + 1) * LDP + kl] = f32_to_bf16_rne(rbp[p].y);
      sB[buf][(bn + 2) * LDP + kl] = f32_to_bf16_rne(rbp[p].z);
      sB[buf][(bn + 3) * LDP + kl] = f32_to_bf16_rne(rbp[p].w);
    }
  };

  load_g(0);
  store_l(0);
  __syncthreads();

  int cur = 0;
  for (int k0 = 0; k0 < CDIM; k0 += 32) {
    const int nxt = k0 + 32;
    if (nxt < CDIM) load_g(nxt);

    bf16x16 af[4], bf[2];
#pragma unroll
    for (int mt = 0; mt < 4; ++mt) {
      const unsigned short* pr = &sA[cur][(wM * 64 + mt * 16 + lane16) * LDP + hp * 8];
      af[mt] = make_frag(pr, pr + 16);
    }
#pragma unroll
    for (int nt = 0; nt < 2; ++nt) {
      const unsigned short* pr = &sB[cur][(wN * 32 + nt * 16 + lane16) * LDP + hp * 8];
      bf[nt] = make_frag(pr, pr + 16);
    }
#pragma unroll
    for (int mt = 0; mt < 4; ++mt)
#pragma unroll
      for (int nt = 0; nt < 2; ++nt)
        acc[mt][nt] = __builtin_amdgcn_wmma_f32_16x16x32_bf16(
            false, af[mt], false, bf[nt], (short)0, acc[mt][nt], false, false);

    if (nxt < CDIM) {
      store_l(cur ^ 1);
      __syncthreads();
      cur ^= 1;
    }
  }

  float* ob = out + (size_t)b * CDIM * HW;
#pragma unroll
  for (int mt = 0; mt < 4; ++mt)
#pragma unroll
    for (int nt = 0; nt < 2; ++nt)
#pragma unroll
      for (int v = 0; v < 8; ++v) {
        const int m = m0 + wM * 64 + mt * 16 + v + 8 * hp;
        const int n = n0 + wN * 32 + nt * 16 + lane16;
        ob[(size_t)m * HW + n] = acc[mt][nt][v];
      }
}

// ---------------------------------------------------------------------------
extern "C" void kernel_launch(void* const* d_in, const int* in_sizes, int n_in,
                              void* d_out, int out_size, void* d_ws, size_t ws_size,
                              hipStream_t stream) {
  (void)in_sizes; (void)n_in; (void)out_size; (void)ws_size;
  const float* x = (const float*)d_in[0];
  float* out = (float*)d_out;

  float*          S = (float*)d_ws;                                        // 64 MB
  unsigned short* P = (unsigned short*)((char*)d_ws +
                      (size_t)BATCH * CDIM * CDIM * sizeof(float));        // +32 MB

  gemm1_scores<<<dim3(CDIM / 128, CDIM / 128, BATCH), 256, 0, stream>>>(x, S);
  softmax_rows<<<BATCH * CDIM, 256, 0, stream>>>(S, P);
  gemm2_out<<<dim3(HW / 128, CDIM / 128, BATCH), 256, 0, stream>>>(P, x, out);
}
